// SparseCodingModel_67705864454780
// MI455X (gfx1250) — matrix operand order repaired and checked by reading, not compiled, not run on hardware
//
#include <hip/hip_runtime.h>
#include <hip/hip_bf16.h>

// ---------------------------------------------------------------------------
// Sparse coding model on MI455X (gfx1250):
//   conv1 (implicit GEMM, bf16 WMMA, fused pass-0 histogram) ->
//   exact top-k (radix select) -> bitonic sort -> deterministic gather recon.
// ---------------------------------------------------------------------------

#define NS   32768   // samples
#define NA   1024    // atoms
#define AL   2048    // atom length
#define KTK  1024    // n_to_keep
#define NB   2       // batches

typedef __attribute__((ext_vector_type(16))) __bf16 v16bf;
typedef __attribute__((ext_vector_type(8)))  float  v8f;

struct Entry { float v; unsigned idx; };

__device__ __forceinline__ unsigned short f2bf(float f) {
    unsigned u = __float_as_uint(f);
    u += 0x7FFFu + ((u >> 16) & 1u);          // round-to-nearest-even
    return (unsigned short)(u >> 16);
}

__device__ __forceinline__ unsigned fkey(float f) {
    // monotone map: larger float -> larger unsigned key
    unsigned u = __float_as_uint(f);
    return (u & 0x80000000u) ? ~u : (u | 0x80000000u);
}

// ---------------------------------------------------------------------------
// 0) setup: bf16 conversion + zero selection state
// ---------------------------------------------------------------------------
__global__ void sc_setup(const float* __restrict__ atoms, const float* __restrict__ x,
                         unsigned short* __restrict__ atomsB, unsigned short* __restrict__ xB,
                         unsigned* __restrict__ hist, unsigned* __restrict__ state,
                         unsigned* __restrict__ cnt, Entry* __restrict__ ent) {
    size_t i = (size_t)blockIdx.x * blockDim.x + threadIdx.x;
    size_t stride = (size_t)gridDim.x * blockDim.x;
    for (size_t j = i; j < (size_t)NA * AL; j += stride) atomsB[j] = f2bf(atoms[j]);
    for (size_t j = i; j < (size_t)NB * NS; j += stride) xB[j]     = f2bf(x[j]);
    for (size_t j = i; j < NB * 4 * 256;   j += stride) hist[j] = 0u;
    for (size_t j = i; j < NB * 2;         j += stride) cnt[j]  = 0u;
    for (size_t j = i; j < NB;             j += stride) { state[j*4+0] = 0u; state[j*4+1] = KTK; }
    for (size_t j = i; j < (size_t)NB*KTK; j += stride) { ent[j].v = 0.f; ent[j].idx = 0u; }
}

// ---------------------------------------------------------------------------
// 1) conv1 as implicit GEMM:  fm[b] = atoms(1024x2048) * Toeplitz(x)(2048x32768)
//    block: 128 atoms x 128 samples, 8 waves, each wave 4x2 WMMA tiles.
//    x-window stored REVERSED in LDS so every B fragment is a forward
//    consecutive run (wide ds loads, no v_mov_b16 reversal).
//    Epilogue: fused radix-select pass-0 histogram of this block's fm tile.
// ---------------------------------------------------------------------------
__global__ __launch_bounds__(256)
void sc_conv1_wmma(const unsigned short* __restrict__ atomsB,
                   const unsigned short* __restrict__ xB,
                   float* __restrict__ fm,
                   unsigned* __restrict__ hist) {
    __shared__ unsigned short rw[2176];   // reversed: rw[i] = x[t0 + 128 - i]
    __shared__ unsigned lh[256];          // pass-0 histogram

    const int t0 = blockIdx.x * 128;      // sample tile
    const int m0 = blockIdx.y * 128;      // atom tile
    const int b  = blockIdx.z;

    lh[threadIdx.x] = 0u;
    const unsigned short* xb = xB + (size_t)b * NS;
    for (int i = threadIdx.x; i < 2176; i += 256) {
        int g = t0 + 128 - i;
        rw[i] = (g >= 0 && g < NS) ? xb[g] : (unsigned short)0;
    }
    __syncthreads();

    const int lane = threadIdx.x & 31;
    const int wave = threadIdx.x >> 5;
    const int r    = lane & 15;
    const int half = lane >> 4;
    const int miB  = (wave >> 2) * 4;     // 0 or 4  (16-row tiles)
    const int niB  = (wave & 3) * 2;      // 0,2,4,6 (16-col tiles)

    v8f acc[4][2];
    #pragma unroll
    for (int i = 0; i < 4; ++i)
        #pragma unroll
        for (int j = 0; j < 2; ++j)
            #pragma unroll
            for (int e = 0; e < 8; ++e) acc[i][j][e] = 0.f;

    const __bf16* rwb = reinterpret_cast<const __bf16*>(rw);
    const __bf16* Ab  = reinterpret_cast<const __bf16*>(atomsB);

    for (int k0 = 0; k0 < AL; k0 += 32) {
        // ---- A fragments (16x32 bf16, contiguous k order, same as B):
        //      lane<16 holds K {0..15}, lane>=16 holds K {16..31}
        v16bf afr[4];
        #pragma unroll
        for (int mi = 0; mi < 4; ++mi) {
            const int row = m0 + (miB + mi) * 16 + r;
            const __bf16* ap = Ab + (size_t)row * AL + k0 + half * 16;
            if (mi == 0) __builtin_prefetch(ap + 256, 0, 0);   // global_prefetch_b8
            #pragma unroll
            for (int e = 0; e < 16; ++e) afr[mi][e] = ap[e];
        }
        // ---- B fragments: B[k, cc] = x[t0+cc-k] = rw[128 - cc + k]
        //      forward consecutive in k -> wide LDS loads
        v16bf bfr[2];
        #pragma unroll
        for (int ni = 0; ni < 2; ++ni) {
            const int cc = (niB + ni) * 16 + r;            // local column
            const __bf16* bp = rwb + (128 - cc + k0 + half * 16);
            #pragma unroll
            for (int e = 0; e < 16; ++e) bfr[ni][e] = bp[e];
        }
        // ---- 8 WMMAs ----
        #pragma unroll
        for (int mi = 0; mi < 4; ++mi)
            #pragma unroll
            for (int ni = 0; ni < 2; ++ni)
                acc[mi][ni] = __builtin_amdgcn_wmma_f32_16x16x32_bf16(
                    false, afr[mi], false, bfr[ni], (short)0, acc[mi][ni], false, false);
    }

    // ---- store C tiles + fused pass-0 histogram
    float* fmB = fm + (size_t)b * NA * NS;
    #pragma unroll
    for (int mi = 0; mi < 4; ++mi)
        #pragma unroll
        for (int ni = 0; ni < 2; ++ni) {
            const int tileM = m0 + (miB + mi) * 16 + half * 8;
            const int tileN = t0 + (niB + ni) * 16 + r;
            #pragma unroll
            for (int rv = 0; rv < 8; ++rv) {
                const float v = acc[mi][ni][rv];
                fmB[(size_t)(tileM + rv) * NS + tileN] = v;
                atomicAdd(&lh[fkey(v) >> 24], 1u);
            }
        }
    __syncthreads();
    atomicAdd(&hist[(b * 4 + 0) * 256 + threadIdx.x], lh[threadIdx.x]);
}

// ---------------------------------------------------------------------------
// 2) radix select: passes 1..3 (pass 0 fused into conv kernel)
// ---------------------------------------------------------------------------
__global__ __launch_bounds__(256)
void sc_hist(const float* __restrict__ fm, unsigned* __restrict__ hist,
             const unsigned* __restrict__ state, int pass) {
    __shared__ unsigned lh[256];
    const int b = blockIdx.y;
    const unsigned prefix = state[b * 4 + 0];
    lh[threadIdx.x] = 0u;
    __syncthreads();

    const float* f = fm + (size_t)b * NA * NS;
    const size_t n = (size_t)NA * NS;
    const int shift = 24 - 8 * pass;
    for (size_t i = (size_t)blockIdx.x * 256 + threadIdx.x; i < n; i += (size_t)gridDim.x * 256) {
        unsigned key = fkey(f[i]);
        if ((key >> (shift + 8)) == prefix)
            atomicAdd(&lh[(key >> shift) & 255u], 1u);
    }
    __syncthreads();
    atomicAdd(&hist[(b * 4 + pass) * 256 + threadIdx.x], lh[threadIdx.x]);
}

__global__ void sc_pick(const unsigned* __restrict__ hist, unsigned* __restrict__ state, int pass) {
    const int b = blockIdx.x;
    if (threadIdx.x != 0) return;
    const unsigned* h = hist + (b * 4 + pass) * 256;
    unsigned remaining = state[b * 4 + 1];
    unsigned cum = 0; int bin = 0;
    for (int i = 255; i >= 0; --i) {
        unsigned c = h[i];
        if (cum + c >= remaining) { bin = i; break; }
        cum += c;
    }
    state[b * 4 + 0] = (state[b * 4 + 0] << 8) | (unsigned)bin;
    state[b * 4 + 1] = remaining - cum;   // quota among keys == threshold
}

// ---------------------------------------------------------------------------
// 3) compaction: exactly KTK entries per batch
// ---------------------------------------------------------------------------
__global__ __launch_bounds__(256)
void sc_compact(const float* __restrict__ fm, const unsigned* __restrict__ state,
                unsigned* __restrict__ cnt, Entry* __restrict__ ent) {
    const int b = blockIdx.y;
    const unsigned T     = state[b * 4 + 0];
    const unsigned quota = state[b * 4 + 1];
    const unsigned nGT   = KTK - quota;
    const float* f = fm + (size_t)b * NA * NS;
    Entry* eb = ent + b * KTK;
    unsigned* cb = cnt + b * 2;
    const size_t n = (size_t)NA * NS;
    for (size_t i = (size_t)blockIdx.x * blockDim.x + threadIdx.x; i < n;
         i += (size_t)gridDim.x * blockDim.x) {
        float v = f[i];
        unsigned key = fkey(v);
        if (key > T) {
            unsigned s = atomicAdd(&cb[0], 1u);
            if (s < KTK) { eb[s].v = v; eb[s].idx = (unsigned)i; }
        } else if (key == T) {
            unsigned s = atomicAdd(&cb[1], 1u);
            if (s < quota) { eb[nGT + s].v = v; eb[nGT + s].idx = (unsigned)i; }
        }
    }
}

// ---------------------------------------------------------------------------
// 4) bitonic sort by index -> deterministic reconstruction order
// ---------------------------------------------------------------------------
__global__ __launch_bounds__(1024)
void sc_sort(Entry* __restrict__ ent) {
    __shared__ float    sv[KTK];
    __shared__ unsigned si[KTK];
    Entry* eb = ent + blockIdx.x * KTK;
    const int t = threadIdx.x;
    sv[t] = eb[t].v; si[t] = eb[t].idx;
    __syncthreads();
    for (int k = 2; k <= KTK; k <<= 1)
        for (int j = k >> 1; j > 0; j >>= 1) {
            int ixj = t ^ j;
            if (ixj > t) {
                bool up = ((t & k) == 0);
                bool swp = up ? (si[t] > si[ixj]) : (si[t] < si[ixj]);
                if (swp) {
                    unsigned ti = si[t]; si[t] = si[ixj]; si[ixj] = ti;
                    float    tv = sv[t]; sv[t] = sv[ixj]; sv[ixj] = tv;
                }
            }
            __syncthreads();
        }
    eb[t].v = sv[t]; eb[t].idx = si[t];
}

// ---------------------------------------------------------------------------
// 5) reconstruction: out[b,t] = sum over entries v * atom[ch, t-p]  (fp32)
// ---------------------------------------------------------------------------
__global__ __launch_bounds__(256)
void sc_recon(const Entry* __restrict__ ent, const float* __restrict__ atoms,
              float* __restrict__ out) {
    __shared__ float    ev[KTK];
    __shared__ unsigned ei[KTK];
    const int b = blockIdx.y;
    const Entry* eb = ent + b * KTK;
    for (int i = threadIdx.x; i < KTK; i += 256) { ev[i] = eb[i].v; ei[i] = eb[i].idx; }
    __syncthreads();
    const int t = blockIdx.x * 256 + threadIdx.x;
    float acc = 0.f;
    for (int i = 0; i < KTK; ++i) {
        const unsigned idx = ei[i];
        const int ch = (int)(idx >> 15);
        const int p  = (int)(idx & 0x7FFFu);
        const int d  = t - p;
        if (d >= 0 && d < AL) acc += ev[i] * atoms[(size_t)ch * AL + d];
    }
    out[(size_t)b * NS + t] = acc;
}

// ---------------------------------------------------------------------------
extern "C" void kernel_launch(void* const* d_in, const int* in_sizes, int n_in,
                              void* d_out, int out_size, void* d_ws, size_t ws_size,
                              hipStream_t stream) {
    const float* x     = (const float*)d_in[0];   // (2,1,32768) f32
    const float* atoms = (const float*)d_in[1];   // (1,1024,2048) f32
    float* out = (float*)d_out;                   // (2,1,32768) f32

    char* ws = (char*)d_ws;
    size_t off = 0;
    auto carve = [&](size_t bytes) { size_t o = off; off = (off + bytes + 255) & ~(size_t)255; return o; };

    unsigned short* atomsB = (unsigned short*)(ws + carve((size_t)NA * AL * 2));
    unsigned short* xB     = (unsigned short*)(ws + carve((size_t)NB * NS * 2));
    float*          fm     = (float*)        (ws + carve((size_t)NB * NA * NS * 4));
    unsigned*       hist   = (unsigned*)     (ws + carve((size_t)NB * 4 * 256 * 4));
    unsigned*       state  = (unsigned*)     (ws + carve((size_t)NB * 4 * 4));
    unsigned*       cnt    = (unsigned*)     (ws + carve((size_t)NB * 2 * 4));
    Entry*          ent    = (Entry*)        (ws + carve((size_t)NB * KTK * sizeof(Entry)));
    (void)ws_size; (void)in_sizes; (void)n_in; (void)out_size;

    // 0) conversions + state init
    sc_setup<<<2048, 256, 0, stream>>>(atoms, x, atomsB, xB, hist, state, cnt, ent);

    // 1) conv1: fm = atoms (*) x  via bf16 WMMA implicit GEMM (+ pass-0 hist)
    sc_conv1_wmma<<<dim3(NS / 128, NA / 128, NB), 256, 0, stream>>>(atomsB, xB, fm, hist);

    // 2) exact top-k threshold: radix select (pass 0 already histogrammed)
    sc_pick<<<NB, 32, 0, stream>>>(hist, state, 0);
    for (int p = 1; p < 4; ++p) {
        sc_hist<<<dim3(1024, NB), 256, 0, stream>>>(fm, hist, state, p);
        sc_pick<<<NB, 32, 0, stream>>>(hist, state, p);
    }

    // 3) gather the KTK winners
    sc_compact<<<dim3(1024, NB), 256, 0, stream>>>(fm, state, cnt, ent);

    // 4) deterministic ordering
    sc_sort<<<NB, 1024, 0, stream>>>(ent);

    // 5) reconstruction (fp32, no atomics)
    sc_recon<<<dim3(NS / 256, NB), 256, 0, stream>>>(ent, atoms, out);
}
